// SE3Transformer_14053132993024
// MI455X (gfx1250) — matrix operand order repaired and checked by reading, not compiled, not run on hardware
//
#include <hip/hip_runtime.h>
#include <math.h>

#define NN 4096
#define EE 65536
#define RMID 32

typedef __attribute__((ext_vector_type(16))) _Float16 v16h;
typedef __attribute__((ext_vector_type(8)))  float    v8f;

// ---------------------------------------------------------------------------
// Fused per-edge kernel: radial MLP (WMMA f16, f32 accum) + basis contraction
// One wave32 block handles a tile of 16 edges. EXEC is all-ones at WMMA sites.
// ---------------------------------------------------------------------------
struct ConvArgs {
  const float* feat;     // [E,2]
  const int*   src;      // [E]
  const float* h;        // [N, CI, MI] gathered via src
  const float* basis;    // [E, mo, MI, NF]
  const float* l1w; const float* l1b; const float* ln1g; const float* ln1b;
  const float* l2w; const float* l2b; const float* ln2g; const float* ln2b;
  const float* l3w; const float* l3b;
  float* out;            // [E, co, mo]
  int co; int mo; int accumulate;
};

template<int CI, int MI, int NF>
__global__ void __launch_bounds__(32) conv_kernel(ConvArgs A)
{
  constexpr int DMAX = 4 * CI * NF;      // co <= 4
  __shared__ float    sFeat[16][2];
  __shared__ _Float16 sAh[16][RMID];     // f16 activations (A-fragment source)
  __shared__ float    sH[16][RMID];      // f32 activations
  __shared__ float    sR[16][DMAX];      // radial output tile
  __shared__ float    sHs[16][CI * MI];  // gathered source features

  const int tid  = threadIdx.x;          // 0..31, one wave
  const int e0   = blockIdx.x * 16;
  const int half = (tid >= 16) ? 1 : 0;
  const int l16  = tid & 15;

  // pull this tile's basis block toward the caches (global_prefetch)
  __builtin_prefetch(A.basis + (size_t)e0 * A.mo * MI * NF, 0, 1);

  sFeat[tid >> 1][tid & 1] = A.feat[(size_t)(e0 + (tid >> 1)) * 2 + (tid & 1)];

  // gather h_src tile
  for (int i = tid; i < 16 * CI * MI; i += 32) {
    int ei  = i / (CI * MI);
    int rem = i - ei * (CI * MI);
    int s   = A.src[e0 + ei];
    sHs[ei][rem] = A.h[(size_t)s * (CI * MI) + rem];
  }
  __syncthreads();

  // ---- radial l1 (K=2, scalar) + LN + ReLU -> f16 ----
  if (tid < 16) {
    float f0 = sFeat[tid][0], f1 = sFeat[tid][1];
    float row[RMID]; float mu = 0.f;
    for (int j = 0; j < RMID; ++j) {
      float x = f0 * A.l1w[2 * j] + f1 * A.l1w[2 * j + 1] + A.l1b[j];
      row[j] = x; mu += x;
    }
    mu *= (1.f / RMID);
    float var = 0.f;
    for (int j = 0; j < RMID; ++j) { float d = row[j] - mu; var += d * d; }
    float inv = rsqrtf(var * (1.f / RMID) + 1e-5f);
    for (int j = 0; j < RMID; ++j) {
      float x = (row[j] - mu) * inv * A.ln1g[j] + A.ln1b[j];
      sAh[tid][j] = (_Float16)(x > 0.f ? x : 0.f);
    }
  }
  __syncthreads();

  // ---- radial l2: [16x32]@W2^T via v_wmma_f32_16x16x32_f16 ----
  v16h a;
#pragma unroll
  for (int hh = 0; hh < 16; ++hh) {
    int k = (hh < 8 ? hh : 16 + (hh - 8)) + (half ? 8 : 0);   // A 16x32 f16 layout
    a[hh] = sAh[l16][k];
  }
#pragma unroll
  for (int nt = 0; nt < 2; ++nt) {
    int n = nt * 16 + l16;
    v16h b;
#pragma unroll
    for (int hh = 0; hh < 16; ++hh) {
      int k = (hh < 8 ? hh : 16 + (hh - 8)) + (half ? 8 : 0); // B 32x16 f16 layout
      b[hh] = (_Float16)A.l2w[n * RMID + k];
    }
    v8f c = {};
    c = __builtin_amdgcn_wmma_f32_16x16x32_f16(false, a, false, b, (short)0, c, false, false);
#pragma unroll
    for (int g = 0; g < 8; ++g) sH[g + 8 * half][n] = c[g] + A.l2b[n];
  }
  __syncthreads();

  // ---- LN2 + ReLU -> f16 ----
  if (tid < 16) {
    float mu = 0.f;
    for (int j = 0; j < RMID; ++j) mu += sH[tid][j];
    mu *= (1.f / RMID);
    float var = 0.f;
    for (int j = 0; j < RMID; ++j) { float d = sH[tid][j] - mu; var += d * d; }
    float inv = rsqrtf(var * (1.f / RMID) + 1e-5f);
    for (int j = 0; j < RMID; ++j) {
      float x = (sH[tid][j] - mu) * inv * A.ln2g[j] + A.ln2b[j];
      sAh[tid][j] = (_Float16)(x > 0.f ? x : 0.f);
    }
  }
  __syncthreads();

  // ---- radial l3: [16x32]@W3^T -> R tile (Dout = co*CI*NF) ----
#pragma unroll
  for (int hh = 0; hh < 16; ++hh) {
    int k = (hh < 8 ? hh : 16 + (hh - 8)) + (half ? 8 : 0);
    a[hh] = sAh[l16][k];
  }
  const int Dout   = A.co * CI * NF;
  const int ntiles = (Dout + 15) >> 4;
  for (int nt = 0; nt < ntiles; ++nt) {
    int  n  = nt * 16 + l16;
    bool ok = (n < Dout);
    v16h b;
#pragma unroll
    for (int hh = 0; hh < 16; ++hh) {
      int k = (hh < 8 ? hh : 16 + (hh - 8)) + (half ? 8 : 0);
      b[hh] = ok ? (_Float16)A.l3w[n * RMID + k] : (_Float16)0.f;
    }
    v8f c = {};
    c = __builtin_amdgcn_wmma_f32_16x16x32_f16(false, a, false, b, (short)0, c, false, false);
#pragma unroll
    for (int g = 0; g < 8; ++g) {
      if (ok) sR[g + 8 * half][n] = c[g] + A.l3b[n];
    }
  }
  __syncthreads();

  // ---- per-edge equivariant contraction:
  // msg[e,o,m] = sum_{c,f} R[e,o,c,f] * sum_n B[e,m,n,f] * h[e,c,n]
  const int com = A.co * A.mo;
  for (int i = tid; i < 16 * com; i += 32) {
    int ei  = i / com;
    int rem = i - ei * com;
    int o   = rem / A.mo;
    int m   = rem - o * A.mo;
    const float* Be = A.basis + ((size_t)(e0 + ei) * A.mo + m) * (MI * NF);
    float Bl[MI * NF];
#pragma unroll
    for (int x = 0; x < MI * NF; ++x) Bl[x] = Be[x];
    float acc = 0.f;
    for (int cc = 0; cc < CI; ++cc) {
      const float* Rr = &sR[ei][(o * CI + cc) * NF];
#pragma unroll
      for (int f = 0; f < NF; ++f) {
        float s = 0.f;
#pragma unroll
        for (int n2 = 0; n2 < MI; ++n2) s += Bl[n2 * NF + f] * sHs[ei][cc * MI + n2];
        acc += Rr[f] * s;
      }
    }
    float* op = A.out + (size_t)(e0 + ei) * com + rem;
    *op = A.accumulate ? (*op + acc) : acc;
  }
}

static void launch_conv(int ci, int mi, int nf, const ConvArgs& A, hipStream_t s) {
  dim3 g(EE / 16), b(32);
#define CC(CI_, MI_, NF_) if (ci == CI_ && mi == MI_ && nf == NF_) { conv_kernel<CI_, MI_, NF_><<<g, b, 0, s>>>(A); return; }
  CC(1, 1, 1) CC(1, 3, 1) CC(1, 3, 3)
  CC(16, 1, 1) CC(16, 3, 1) CC(16, 3, 3)
  CC(16, 5, 1) CC(16, 5, 3) CC(16, 5, 5)
  CC(16, 7, 1) CC(16, 7, 3) CC(16, 7, 5) CC(16, 7, 7)
#undef CC
}

// ---------------------------------------------------------------------------
// Elementwise / attention kernels
// ---------------------------------------------------------------------------
__global__ void feat_kernel(const float* w, const float* r, float* feat) {
  int e = blockIdx.x * blockDim.x + threadIdx.x;
  if (e < EE) { feat[2 * e] = w[e]; feat[2 * e + 1] = r[e]; }
}
__global__ void zero_kernel(float* p, int n) {
  int i = blockIdx.x * blockDim.x + threadIdx.x;
  if (i < n) p[i] = 0.f;
}
__global__ void g1x1_kernel(const float* w, const float* h, float* out, int co, int ci, int m) {
  int i = blockIdx.x * blockDim.x + threadIdx.x;
  int tot = NN * co * m;
  if (i >= tot) return;
  int n = i / (co * m); int rem = i - n * (co * m); int o = rem / m; int mm = rem - o * m;
  float acc = 0.f;
  for (int c = 0; c < ci; ++c) acc += w[o * ci + c] * h[((size_t)n * ci + c) * m + mm];
  out[((size_t)n * co + o) * m + mm] = acc;
}
__global__ void proj_kernel(const float* w, const float* z, int cz, const float* hin, int ci,
                            float* out, int co, int m) {
  int i = blockIdx.x * blockDim.x + threadIdx.x;
  int tot = NN * co * m;
  if (i >= tot) return;
  int n = i / (co * m); int rem = i - n * (co * m); int o = rem / m; int mm = rem - o * m;
  int ct = cz + ci;
  float acc = 0.f;
  for (int c = 0; c < cz; ++c) acc += w[o * ct + c] * z[((size_t)n * cz + c) * m + mm];
  for (int c = 0; c < ci; ++c) acc += w[o * ct + cz + c] * hin[((size_t)n * ci + c) * m + mm];
  out[((size_t)n * co + o) * m + mm] = acc;
}

__device__ __forceinline__ unsigned fkey(float f) {
  unsigned u = __float_as_uint(f);
  return (u & 0x80000000u) ? ~u : (u | 0x80000000u);
}
__device__ __forceinline__ float funkey(unsigned k) {
  unsigned u = (k & 0x80000000u) ? (k & 0x7fffffffu) : ~k;
  return __uint_as_float(u);
}

struct ScoreArgs {
  const float* ke[4]; const float* qn[4]; int cm[4]; int nd;
  float scale; const int* dst; float* score; unsigned* maxK;
};
__global__ void score_kernel(ScoreArgs A) {
  int e = blockIdx.x * blockDim.x + threadIdx.x;
  if (e >= EE) return;
  int d0 = A.dst[e];
  float acc = 0.f;
  for (int d = 0; d < A.nd; ++d) {
    int cm = A.cm[d];
    const float* kp = A.ke[d] + (size_t)e * cm;
    const float* qp = A.qn[d] + (size_t)d0 * cm;
    for (int j = 0; j < cm; ++j) acc += kp[j] * qp[j];
  }
  float s = acc * A.scale;
  A.score[e] = s;
  atomicMax(A.maxK + d0, fkey(s));
}
__global__ void den_kernel(const float* score, const int* dst, const unsigned* maxK, float* den) {
  int e = blockIdx.x * blockDim.x + threadIdx.x;
  if (e >= EE) return;
  int d0 = dst[e];
  atomicAdd(den + d0, __expf(score[e] - funkey(maxK[d0])));
}
struct ScatArgs {
  const float* ve[4]; float* z[4]; int cm[4]; int nd;
  const int* dst; const float* score; const unsigned* maxK; const float* den;
};
__global__ void scatter_kernel(ScatArgs A) {
  int e = blockIdx.x * blockDim.x + threadIdx.x;
  if (e >= EE) return;
  int d0 = A.dst[e];
  float dn = A.den[d0]; dn = dn > 1e-12f ? dn : 1e-12f;
  float alpha = __expf(A.score[e] - funkey(A.maxK[d0])) / dn;
  for (int d = 0; d < A.nd; ++d) {
    int cm = A.cm[d];
    const float* vp = A.ve[d] + (size_t)e * cm;
    float* zp = A.z[d] + (size_t)d0 * cm;
    for (int j = 0; j < cm; ++j) atomicAdd(zp + j, alpha * vp[j]);
  }
}
__global__ void normbias_kernel(float* h, const float* bias, int c, int m) {
  int i = blockIdx.x * blockDim.x + threadIdx.x;
  int tot = NN * c;
  if (i >= tot) return;
  int n = i / c; int cc = i - n * c;
  float* x = h + ((size_t)n * c + cc) * m;
  float s = 1e-12f;
  for (int j = 0; j < m; ++j) s += x[j] * x[j];
  float nr = sqrtf(s);
  float t = nr + bias[cc]; t = t > 0.f ? t : 0.f;
  float sc = t / nr;
  for (int j = 0; j < m; ++j) x[j] *= sc;
}

// ---------------------------------------------------------------------------
// Host-side parameter-index resolution (handles insertion-order and
// jax-sorted-key pytree flattening; auto-detected from in_sizes[0]).
// ---------------------------------------------------------------------------
struct RadialIdx { int l1w, l1b, ln1g, ln1b, l2w, l2b, ln2g, ln2b, l3w, l3b; };
struct ResIdx { RadialIdx v[16]; RadialIdx k[16]; int q[4]; int proj[4]; };
struct ResCfg {
  int ndi; int di[4]; int ci;
  int nvdo; int vdo[4]; int vco;
  int nkdo; int kdo[4]; int kco;
  int nprojd; int projd[4]; int projco;
};

static void walk_radial(int& cur, int scheme, RadialIdx& R) {
  if (scheme == 0) { // insertion: l1{w,b} ln1{g,b} l2{w,b} ln2{g,b} l3{w,b}
    R.l1w = cur++; R.l1b = cur++; R.ln1g = cur++; R.ln1b = cur++;
    R.l2w = cur++; R.l2b = cur++; R.ln2g = cur++; R.ln2b = cur++;
    R.l3w = cur++; R.l3b = cur++;
  } else {           // sorted: l1{b,w} l2{b,w} l3{b,w} ln1{b,g} ln2{b,g}
    R.l1b = cur++; R.l1w = cur++; R.l2b = cur++; R.l2w = cur++;
    R.l3b = cur++; R.l3w = cur++;
    R.ln1b = cur++; R.ln1g = cur++; R.ln2b = cur++; R.ln2g = cur++;
  }
}
static void walk_res(int& cur, int scheme, ResIdx& R, const ResCfg& C) {
  int nv = C.ndi * C.nvdo, nk = C.ndi * C.nkdo;
  if (scheme == 0) { // v, k, q, proj
    for (int i = 0; i < nv; ++i) walk_radial(cur, scheme, R.v[i]);
    for (int i = 0; i < nk; ++i) walk_radial(cur, scheme, R.k[i]);
    for (int i = 0; i < C.nkdo; ++i) R.q[i] = cur++;
    for (int i = 0; i < C.nprojd; ++i) R.proj[i] = cur++;
  } else {           // k, proj, q, v
    for (int i = 0; i < nk; ++i) walk_radial(cur, scheme, R.k[i]);
    for (int i = 0; i < C.nprojd; ++i) R.proj[i] = cur++;
    for (int i = 0; i < C.nkdo; ++i) R.q[i] = cur++;
    for (int i = 0; i < nv; ++i) walk_radial(cur, scheme, R.v[i]);
  }
}

struct Ws {
  float* feat; float* hA; float* hB; float* ve; float* ke; float* qn;
  float* score; unsigned* maxK; float* den; float* z;
};

static void run_res(void* const* d_in, int basis_base,
                    const ResIdx& RI, const ResCfg& C,
                    const int* src, const int* dst,
                    const float* const* hin, int hci,
                    float* houtBase, float* finalOut,
                    const int* normRow,
                    const Ws& W, hipStream_t s)
{
  auto P = [&](int i) { return (const float*)d_in[i]; };
  size_t veOff[4], zOff[4], keOff[4], qnOff[4];
  size_t o;
  o = 0; for (int i = 0; i < C.nvdo; ++i) { veOff[i] = o; o += (size_t)EE * C.vco * (2 * C.vdo[i] + 1); }
  o = 0; for (int i = 0; i < C.nvdo; ++i) { zOff[i]  = o; o += (size_t)NN * C.vco * (2 * C.vdo[i] + 1); }
  size_t ztot = o;
  o = 0; for (int i = 0; i < C.nkdo; ++i) { keOff[i] = o; o += (size_t)EE * C.kco * (2 * C.kdo[i] + 1); }
  o = 0; for (int i = 0; i < C.nkdo; ++i) { qnOff[i] = o; o += (size_t)NN * C.kco * (2 * C.kdo[i] + 1); }

  // V and K edge convolutions (fused radial MLP + basis contraction)
  for (int b = 0; b < 2; ++b) {
    int ndo        = b ? C.nkdo : C.nvdo;
    const int* dol = b ? C.kdo  : C.vdo;
    int co         = b ? C.kco  : C.vco;
    float* obuf    = b ? W.ke   : W.ve;
    const size_t* ooff = b ? keOff : veOff;
    for (int oi = 0; oi < ndo; ++oi) {
      int dd = dol[oi], mo = 2 * dd + 1;
      for (int ti = 0; ti < C.ndi; ++ti) {
        int di = C.di[ti], mi = 2 * di + 1;
        int mn = di < dd ? di : dd;
        int nf = 2 * mn + 1;
        const RadialIdx& RR = b ? RI.k[ti * C.nkdo + oi] : RI.v[ti * C.nvdo + oi];
        ConvArgs A;
        A.feat = W.feat; A.src = src;
        A.h = hin[di];
        A.basis = P(basis_base + di * 4 + dd);
        A.l1w = P(RR.l1w); A.l1b = P(RR.l1b); A.ln1g = P(RR.ln1g); A.ln1b = P(RR.ln1b);
        A.l2w = P(RR.l2w); A.l2b = P(RR.l2b); A.ln2g = P(RR.ln2g); A.ln2b = P(RR.ln2b);
        A.l3w = P(RR.l3w); A.l3b = P(RR.l3b);
        A.out = obuf + ooff[oi]; A.co = co; A.mo = mo; A.accumulate = (ti > 0);
        launch_conv(C.ci, mi, nf, A, s);
      }
    }
  }
  // Q projections over nodes
  for (int oi = 0; oi < C.nkdo; ++oi) {
    int dd = C.kdo[oi], m = 2 * dd + 1;
    int tot = NN * C.kco * m;
    g1x1_kernel<<<(tot + 255) / 256, 256, 0, s>>>(P(RI.q[oi]), hin[dd], W.qn + qnOff[oi], C.kco, hci, m);
  }
  // zero maxK / den / z (contiguous region)
  {
    int tot = (int)((size_t)2 * NN + ztot);
    zero_kernel<<<(tot + 255) / 256, 256, 0, s>>>((float*)W.maxK, tot);
  }
  // edge scores + segment softmax + weighted scatter
  {
    ScoreArgs SA{};
    SA.nd = C.nkdo; int dim = 0;
    for (int oi = 0; oi < C.nkdo; ++oi) {
      SA.ke[oi] = W.ke + keOff[oi];
      SA.qn[oi] = W.qn + qnOff[oi];
      SA.cm[oi] = C.kco * (2 * C.kdo[oi] + 1);
      dim += SA.cm[oi];
    }
    SA.scale = 1.0f / sqrtf((float)dim);
    SA.dst = dst; SA.score = W.score; SA.maxK = W.maxK;
    score_kernel<<<EE / 256, 256, 0, s>>>(SA);
    den_kernel<<<EE / 256, 256, 0, s>>>(W.score, dst, W.maxK, W.den);
    ScatArgs SC{};
    SC.nd = C.nvdo;
    for (int oi = 0; oi < C.nvdo; ++oi) {
      SC.ve[oi] = W.ve + veOff[oi];
      SC.z[oi]  = W.z + zOff[oi];
      SC.cm[oi] = C.vco * (2 * C.vdo[oi] + 1);
    }
    SC.dst = dst; SC.score = W.score; SC.maxK = W.maxK; SC.den = W.den;
    scatter_kernel<<<EE / 256, 256, 0, s>>>(SC);
  }
  // concat-skip + output projection (+ optional GNormBias)
  const size_t hOff[4] = { 0, (size_t)16 * NN, (size_t)64 * NN, (size_t)144 * NN };
  for (int pi = 0; pi < C.nprojd; ++pi) {
    int dd = C.projd[pi], m = 2 * dd + 1;
    int zi = 0; for (int oi = 0; oi < C.nvdo; ++oi) if (C.vdo[oi] == dd) zi = oi;
    const float* hp = hin[dd];
    int cip = hp ? hci : 0;
    float* outp = finalOut ? finalOut : (houtBase + hOff[dd]);
    int tot = NN * C.projco * m;
    proj_kernel<<<(tot + 255) / 256, 256, 0, s>>>(P(RI.proj[pi]), W.z + zOff[zi], C.vco, hp, cip,
                                                  outp, C.projco, m);
    if (normRow) {
      int tot2 = NN * C.projco;
      normbias_kernel<<<(tot2 + 255) / 256, 256, 0, s>>>(outp, P(normRow[dd]), C.projco, m);
    }
  }
}

extern "C" void kernel_launch(void* const* d_in, const int* in_sizes, int n_in,
                              void* d_out, int out_size, void* d_ws, size_t ws_size,
                              hipStream_t stream)
{
  (void)n_in; (void)out_size; (void)ws_size;
  // scheme 0: top-level insertion order (c first, 4096 elems)
  // scheme 1: jax sorted-key flatten (basis['0,0'] first, 65536 elems)
  int scheme = (in_sizes[0] == NN) ? 0 : 1;
  int idx_c, idx_v, idx_w, idx_r, idx_src, idx_dst, basis_base, params_base;
  if (scheme == 0) {
    idx_c = 0; idx_v = 1; idx_w = 2; idx_r = 3; basis_base = 4; params_base = 20;
    idx_src = 1228; idx_dst = 1229;
  } else {
    basis_base = 0; idx_c = 16; idx_dst = 17; params_base = 18;
    idx_r = 1226; idx_src = 1227; idx_v = 1228; idx_w = 1229;
  }

  ResCfg cfg[5];
  cfg[0].ndi = 2; cfg[0].di[0] = 0; cfg[0].di[1] = 1; cfg[0].di[2] = 0; cfg[0].di[3] = 0; cfg[0].ci = 1;
  cfg[0].nvdo = 4; for (int i = 0; i < 4; ++i) cfg[0].vdo[i] = i; cfg[0].vco = 4;
  cfg[0].nkdo = 2; cfg[0].kdo[0] = 0; cfg[0].kdo[1] = 1; cfg[0].kdo[2] = 0; cfg[0].kdo[3] = 0; cfg[0].kco = 4;
  cfg[0].nprojd = 4; for (int i = 0; i < 4; ++i) cfg[0].projd[i] = i; cfg[0].projco = 16;
  for (int rr = 1; rr < 4; ++rr) {
    cfg[rr].ndi = 4; for (int i = 0; i < 4; ++i) cfg[rr].di[i] = i; cfg[rr].ci = 16;
    cfg[rr].nvdo = 4; for (int i = 0; i < 4; ++i) cfg[rr].vdo[i] = i; cfg[rr].vco = 4;
    cfg[rr].nkdo = 4; for (int i = 0; i < 4; ++i) cfg[rr].kdo[i] = i; cfg[rr].kco = 4;
    cfg[rr].nprojd = 4; for (int i = 0; i < 4; ++i) cfg[rr].projd[i] = i; cfg[rr].projco = 16;
  }
  cfg[4].ndi = 4; for (int i = 0; i < 4; ++i) cfg[4].di[i] = i; cfg[4].ci = 16;
  cfg[4].nvdo = 1; cfg[4].vdo[0] = 1; cfg[4].vdo[1] = 0; cfg[4].vdo[2] = 0; cfg[4].vdo[3] = 0; cfg[4].vco = 2;
  cfg[4].nkdo = 1; cfg[4].kdo[0] = 1; cfg[4].kdo[1] = 0; cfg[4].kdo[2] = 0; cfg[4].kdo[3] = 0; cfg[4].kco = 2;
  cfg[4].nprojd = 1; cfg[4].projd[0] = 1; cfg[4].projd[1] = 0; cfg[4].projd[2] = 0; cfg[4].projd[3] = 0;
  cfg[4].projco = 2;

  ResIdx res[5]; int norm[4][4];
  int cur = params_base;
  if (scheme == 0) {
    for (int i = 0; i < 4; ++i) { walk_res(cur, 0, res[i], cfg[i]); for (int d = 0; d < 4; ++d) norm[i][d] = cur++; }
    walk_res(cur, 0, res[4], cfg[4]);
  } else {
    for (int i = 0; i < 4; ++i) for (int d = 0; d < 4; ++d) norm[i][d] = cur++;
    for (int i = 0; i < 5; ++i) walk_res(cur, 1, res[i], cfg[i]);
  }

  // workspace carve-out (~45 MB)
  float* wp = (float*)d_ws;
  Ws W; size_t o = 0;
  W.feat  = wp + o; o += (size_t)2 * EE;
  W.hA    = wp + o; o += (size_t)256 * NN;
  W.hB    = wp + o; o += (size_t)256 * NN;
  W.ve    = wp + o; o += (size_t)64 * EE;
  W.ke    = wp + o; o += (size_t)64 * EE;
  W.qn    = wp + o; o += (size_t)64 * NN;
  W.score = wp + o; o += (size_t)EE;
  W.maxK  = (unsigned*)(wp + o); o += (size_t)NN;
  W.den   = wp + o; o += (size_t)NN;
  W.z     = wp + o; o += (size_t)64 * NN;

  const float* c = (const float*)d_in[idx_c];
  const float* v = (const float*)d_in[idx_v];
  const float* w = (const float*)d_in[idx_w];
  const float* r = (const float*)d_in[idx_r];
  const int* src = (const int*)d_in[idx_src];
  const int* dst = (const int*)d_in[idx_dst];

  feat_kernel<<<EE / 256, 256, 0, stream>>>(w, r, W.feat);

  const size_t hOff[4] = { 0, (size_t)16 * NN, (size_t)64 * NN, (size_t)144 * NN };

  // layer 0 (inputs are raw c/v fibers, 1 channel each)
  const float* hin0[4] = { c, v, nullptr, nullptr };
  run_res(d_in, basis_base, res[0], cfg[0], src, dst, hin0, 1, W.hB, nullptr, norm[0], W, stream);

  float* hcur = W.hB; float* hnext = W.hA;
  for (int L = 1; L < 4; ++L) {
    const float* hin[4] = { hcur + hOff[0], hcur + hOff[1], hcur + hOff[2], hcur + hOff[3] };
    run_res(d_in, basis_base, res[L], cfg[L], src, dst, hin, 16, hnext, nullptr, norm[L], W, stream);
    float* t = hcur; hcur = hnext; hnext = t;
  }
  const float* hinF[4] = { hcur + hOff[0], hcur + hOff[1], hcur + hOff[2], hcur + hOff[3] };
  run_res(d_in, basis_base, res[4], cfg[4], src, dst, hinF, 16, nullptr, (float*)d_out, nullptr, W, stream);
}
